// GNNEncoder_15298673509107
// MI455X (gfx1250) — matrix operand order repaired and checked by reading, not compiled, not run on hardware
//
#include <hip/hip_runtime.h>

// ---------------------------------------------------------------------------
// GNN SAGEConv x4 for MI455X (gfx1250, wave32).
//  - edge phase: coalesced gather + global_atomic_add_f32 scatter (L2-resident)
//  - node phase: v_wmma_f32_16x16x32_bf16, 16 nodes x 16 outputs per wave
// ---------------------------------------------------------------------------

typedef __attribute__((ext_vector_type(16))) __bf16 v16bf;
typedef __attribute__((ext_vector_type(8)))  float  v8f;

// ---------------- utility kernels ------------------------------------------

__global__ void zero_f32(float* __restrict__ p, long long n) {
  long long i = (long long)blockIdx.x * blockDim.x + threadIdx.x;
  if (i < n) p[i] = 0.0f;
}

__global__ void degree_kernel(const int* __restrict__ dst, float* __restrict__ deg,
                              int n_edges) {
  int e = blockIdx.x * blockDim.x + threadIdx.x;
  if (e < n_edges) atomicAdd(&deg[dst[e]], 1.0f);
}

__global__ void rdeg_kernel(float* __restrict__ deg, int n) {
  int i = blockIdx.x * blockDim.x + threadIdx.x;
  if (i < n) deg[i] = 1.0f / fmaxf(deg[i], 1.0f);
}

// ---------------- edge scatter ---------------------------------------------
// One thread per (edge, 4-channel chunk). Consecutive D/4 threads share one
// edge -> the random h[src] row is read as one contiguous coalesced burst
// (256B for D=64, 128B for D=32); 4 f32 atomics into summed[dst].

template <int D>
__global__ void __launch_bounds__(256)
scatter_kernel(const float* __restrict__ h, const int* __restrict__ src,
               const int* __restrict__ dst, float* __restrict__ summed, int n_edges) {
  constexpr int CH = D / 4;
  long long tid = (long long)blockIdx.x * blockDim.x + threadIdx.x;
  if (tid >= (long long)n_edges * CH) return;
  int e  = (int)(tid / CH);
  int c4 = (int)(tid % CH) * 4;
  int s = src[e];
  int d = dst[e];
  float4 v = *(const float4*)(h + (size_t)s * D + c4);
  float* p = summed + (size_t)d * D + c4;
  atomicAdd(p + 0, v.x);
  atomicAdd(p + 1, v.y);
  atomicAdd(p + 2, v.z);
  atomicAdd(p + 3, v.w);
}

// ---------------- WMMA tile loaders ----------------------------------------
// A 16x32 bf16: lane L holds row M=L&15; K elements
//   e=0..15 -> K = kb + 16*(L>>4) + e
__device__ __forceinline__ v16bf load_a_tile(const float* __restrict__ row,
                                             int kb, int lane, float scale) {
  const float* r = row + kb + ((lane >> 4) * 16);
  v16bf a;
#pragma unroll
  for (int e = 0; e < 8; ++e) a[e] = (__bf16)(r[e] * scale);
#pragma unroll
  for (int e = 0; e < 8; ++e) a[8 + e] = (__bf16)(r[8 + e] * scale);
  return a;
}

// B 32x16 bf16: lane L holds column n=L&15; element e -> K = kb + 16*(L>>4) + e.
// B[k][n] = W[n][k] (out = in @ W^T), so a lane reads 16 contiguous f32 of W row n.
__device__ __forceinline__ v16bf load_b_tile(const float* __restrict__ W, int din,
                                             int ncol, int kb, int lane) {
  const float* r = W + (size_t)(ncol + (lane & 15)) * din + kb + ((lane >> 4) * 16);
  v16bf b;
#pragma unroll
  for (int e = 0; e < 16; ++e) b[e] = (__bf16)r[e];
  return b;
}

// ---------------- node update: agg@WlT + bl + h@WrT (+skip, relu) -----------
// One wave -> 16 nodes; DOUT/16 column tiles; mean (1/deg) fused into A load.

template <int DIN, int DOUT, bool RELU, bool SKIP>
__global__ void __launch_bounds__(256)
node_wmma_kernel(const float* __restrict__ h, const float* __restrict__ summed,
                 const float* __restrict__ rdeg,
                 const float* __restrict__ Wl, const float* __restrict__ bl,
                 const float* __restrict__ Wr,
                 float* __restrict__ out, int n_nodes) {
  int gtid = blockIdx.x * blockDim.x + threadIdx.x;
  int wave = gtid >> 5;            // wave32
  int base = wave * 16;            // wave-uniform
  if (base >= n_nodes) return;     // uniform branch: EXEC stays all-1s for WMMA
  int lane = threadIdx.x & 31;
  int m = lane & 15;
  float scale = rdeg[base + m];    // mean-aggregation scale for this lane's row
  const float* hrow = h      + (size_t)(base + m) * DIN;
  const float* srow = summed + (size_t)(base + m) * DIN;

#pragma unroll
  for (int ntile = 0; ntile < DOUT / 16; ++ntile) {
    v8f c = {};
#pragma unroll
    for (int kb = 0; kb < DIN; kb += 32) {
      v16bf aA = load_a_tile(srow, kb, lane, scale);
      v16bf bL = load_b_tile(Wl, DIN, ntile * 16, kb, lane);
      c = __builtin_amdgcn_wmma_f32_16x16x32_bf16(false, aA, false, bL,
                                                  (short)0, c, false, false);
      v16bf aH = load_a_tile(hrow, kb, lane, 1.0f);
      v16bf bR = load_b_tile(Wr, DIN, ntile * 16, kb, lane);
      c = __builtin_amdgcn_wmma_f32_16x16x32_bf16(false, aH, false, bR,
                                                  (short)0, c, false, false);
    }
    // C/D layout: VGPR r -> row M = r + 8*(lane>>4), col N = lane&15
    int n = ntile * 16 + (lane & 15);
    float bias = bl[n];
#pragma unroll
    for (int r = 0; r < 8; ++r) {
      int mm = r + 8 * (lane >> 4);
      float v = c[r] + bias;
      if (SKIP) v += h[(size_t)(base + mm) * DIN + n];   // DIN==DOUT on skip layers
      if (RELU) v = fmaxf(v, 0.0f);
      out[(size_t)(base + mm) * DOUT + n] = v;           // lanes 0-15 coalesce per row
    }
  }
}

// ---------------- host launch ----------------------------------------------

extern "C" void kernel_launch(void* const* d_in, const int* in_sizes, int n_in,
                              void* d_out, int out_size, void* d_ws, size_t ws_size,
                              hipStream_t stream) {
  const float* x   = (const float*)d_in[0];
  const int*   ei  = (const int*)d_in[1];
  const int n_nodes = in_sizes[0] / 64;
  const int n_edges = in_sizes[1] / 2;
  const int* src = ei;
  const int* dst = ei + n_edges;

  const float* Wl0 = (const float*)d_in[2];
  const float* bl0 = (const float*)d_in[3];
  const float* Wr0 = (const float*)d_in[4];
  const float* Wl1 = (const float*)d_in[5];
  const float* bl1 = (const float*)d_in[6];
  const float* Wr1 = (const float*)d_in[7];
  const float* Wl2 = (const float*)d_in[8];
  const float* bl2 = (const float*)d_in[9];
  const float* Wr2 = (const float*)d_in[10];
  const float* Wl3 = (const float*)d_in[11];
  const float* bl3 = (const float*)d_in[12];
  const float* Wr3 = (const float*)d_in[13];

  // workspace layout (floats): rdeg[N] | summed[N*64] | hA[N*32] | hB[N*32]
  float* ws     = (float*)d_ws;
  float* rdeg   = ws;
  float* summed = ws + n_nodes;
  float* hA     = summed + (size_t)n_nodes * 64;
  float* hB     = hA + (size_t)n_nodes * 32;
  float* outp   = (float*)d_out;

  const int B = 256;
  auto blk = [](long long n, int b) { return (unsigned int)((n + b - 1) / b); };

  // ---- degree -> 1/max(deg,1) (shared by all layers) ----
  zero_f32<<<blk(n_nodes, B), B, 0, stream>>>(rdeg, n_nodes);
  degree_kernel<<<blk(n_edges, B), B, 0, stream>>>(dst, rdeg, n_edges);
  rdeg_kernel<<<blk(n_nodes, B), B, 0, stream>>>(rdeg, n_nodes);

  const unsigned int node_blocks = blk((n_nodes + 15) / 16, 8);  // 8 waves / 256-thr block
  const long long s64 = (long long)n_nodes * 64;
  const long long s32 = (long long)n_nodes * 32;
  const long long e64 = (long long)n_edges * 16;  // edge*chunk threads, D=64
  const long long e32 = (long long)n_edges * 8;   // D=32

  // ---- layer 0: x(64) -> hA(32), relu, no skip ----
  zero_f32<<<blk(s64, B), B, 0, stream>>>(summed, s64);
  scatter_kernel<64><<<blk(e64, B), B, 0, stream>>>(x, src, dst, summed, n_edges);
  node_wmma_kernel<64, 32, true, false><<<node_blocks, B, 0, stream>>>(
      x, summed, rdeg, Wl0, bl0, Wr0, hA, n_nodes);

  // ---- layer 1: hA -> hB, relu, skip ----
  zero_f32<<<blk(s32, B), B, 0, stream>>>(summed, s32);
  scatter_kernel<32><<<blk(e32, B), B, 0, stream>>>(hA, src, dst, summed, n_edges);
  node_wmma_kernel<32, 32, true, true><<<node_blocks, B, 0, stream>>>(
      hA, summed, rdeg, Wl1, bl1, Wr1, hB, n_nodes);

  // ---- layer 2: hB -> hA, relu, skip ----
  zero_f32<<<blk(s32, B), B, 0, stream>>>(summed, s32);
  scatter_kernel<32><<<blk(e32, B), B, 0, stream>>>(hB, src, dst, summed, n_edges);
  node_wmma_kernel<32, 32, true, true><<<node_blocks, B, 0, stream>>>(
      hB, summed, rdeg, Wl2, bl2, Wr2, hA, n_nodes);

  // ---- layer 3: hA -> out, no relu, no skip ----
  zero_f32<<<blk(s32, B), B, 0, stream>>>(summed, s32);
  scatter_kernel<32><<<blk(e32, B), B, 0, stream>>>(hA, src, dst, summed, n_edges);
  node_wmma_kernel<32, 32, false, false><<<node_blocks, B, 0, stream>>>(
      hA, summed, rdeg, Wl3, bl3, Wr3, outp, n_nodes);
}
